// GAT_41231686042228
// MI455X (gfx1250) — compile-verified, matrix operand order checked by reading, not run on hardware
//
#include <hip/hip_runtime.h>

typedef __attribute__((ext_vector_type(16))) _Float16 v16h;
typedef __attribute__((ext_vector_type(8)))  _Float16 h8;
typedef __attribute__((ext_vector_type(8)))  float    v8f;

#define NNODES 64
#define NCLASS 10
#define EMB    64
#define HID    64
#define HEADS  8
#define DEG    16
#define LEAKYF 0.2f

// padded LDS row lengths (all multiples of 16 bytes, de-phased vs 64 banks)
#define R_H0 72    // halves: h0 / per-head H rows (64 + 8)
#define R_O1 520   // halves: out1 rows (512 + 8)
#define R_O2 68    // floats: out2 rows (64 + 4)

// ---------------------------------------------------------------------------
// Pre-pass: convert W1 (64x512) and W2 (512x64) f32 -> f16, swizzled into the
// WMMA B-fragment lane layout so the GEMM loop does one contiguous v16h load.
// Fragment = 32(K) x 16(N); per lane l: n = l&15, hi = l>>4;
//   half h<8  -> K = 8*hi + h
//   half h>=8 -> K = 16 + 8*hi + (h-8)
// W1 fragments: f = nt*2 + kt  (nt 0..31, kt 0..1)
// W2 fragments: f = nt*16 + kt (nt 0..3,  kt 0..15), stored at +32768 halves
// ---------------------------------------------------------------------------
__global__ void pack_weights(const float* __restrict__ W1,
                             const float* __restrict__ W2,
                             _Float16* __restrict__ wp) {
  int idx = blockIdx.x * blockDim.x + threadIdx.x;
  if (idx >= 65536) return;
  int which = idx >> 15;
  int j  = idx & 32767;
  int f  = j >> 9;
  int r  = j & 511;
  int l  = r >> 4;
  int h  = r & 15;
  int hi = l >> 4;
  int kk = (h < 8) ? (8 * hi + h) : (16 + 8 * hi + (h - 8));
  int nl = l & 15;
  if (which == 0) {
    int nt = f >> 1, kt = f & 1;
    int K = kt * 32 + kk;
    int n = nt * 16 + nl;
    wp[j] = (_Float16)W1[K * (HEADS * HID) + n];
  } else {
    int nt = f >> 4, kt = f & 15;
    int K = kt * 32 + kk;
    int n = nt * 16 + nl;
    wp[32768 + j] = (_Float16)W2[K * HID + n];
  }
}

// ---------------------------------------------------------------------------
// Fused GAT: one workgroup (16 waves) per graph. Everything stays in LDS.
// ---------------------------------------------------------------------------
__global__ __launch_bounds__(512)
void gat_fused(const int*   __restrict__ x,
               const int*   __restrict__ eidx,     // edge_index, row 0 = src
               const float* __restrict__ emb,
               const float* __restrict__ as1,
               const float* __restrict__ ad1,
               const float* __restrict__ b1,
               const float* __restrict__ as2,
               const float* __restrict__ ad2,
               const float* __restrict__ b2,
               const float* __restrict__ headW,
               const float* __restrict__ headb,
               const _Float16* __restrict__ wp,
               float* __restrict__ out) {
  __shared__ __attribute__((aligned(16))) _Float16 sm_h0[NNODES][R_H0];
  __shared__ __attribute__((aligned(16))) _Float16 sm_hk[NNODES][R_H0];  // H1 head-block, later H2
  __shared__ __attribute__((aligned(16))) _Float16 sm_o1[NNODES][R_O1];
  __shared__ __attribute__((aligned(16))) float    sm_o2[NNODES][R_O2];
  __shared__ float sm_w[NNODES][17];
  __shared__ float sm_as[NNODES];
  __shared__ float sm_ad[NNODES];
  __shared__ int   sm_src[NNODES * DEG];

  const int b    = blockIdx.x;
  const int t    = threadIdx.x;
  const int lane = t & 31;
  const int wv   = t >> 5;            // 0..15
  const int hi   = lane >> 4;
  const int nl   = lane & 15;
  const int mt   = wv >> 2;           // output tile row (both GEMMs)
  const int nt   = wv & 3;            // output tile col
  const int m    = mt * 16 + nl;      // A-fragment source row

  const _Float16* w1p = wp;
  const _Float16* w2p = wp + 32768;

  // ---- load local edge sources -------------------------------------------
  for (int i = t; i < NNODES * DEG; i += 512)
    sm_src[i] = eidx[b * NNODES * DEG + i] - b * NNODES;

  // ---- embedding gather -> f16 -------------------------------------------
  {
    int n  = t >> 3;
    int c0 = (t & 7) * 8;
    int xi = x[b * NNODES + n];
    xi = xi < 0 ? 0 : (xi > NCLASS ? NCLASS : xi);
    const float* e = emb + ((size_t)n * (NCLASS + 1) + xi) * EMB + c0;
    #pragma unroll
    for (int i = 0; i < 8; ++i) sm_h0[n][c0 + i] = (_Float16)e[i];
  }
  __syncthreads();

  // ======================= layer 1, head-blocked ==========================
  for (int k = 0; k < HEADS; ++k) {
    // GEMM1 tile: hk(64x64) = h0(64x64) @ W1[:, k*64 : k*64+64]
    {
      v8f acc = {};
      #pragma unroll
      for (int kt = 0; kt < 2; ++kt) {
        const h8 alo = *(const h8*)&sm_h0[m][kt * 32 + 8 * hi];
        const h8 ahi = *(const h8*)&sm_h0[m][kt * 32 + 16 + 8 * hi];
        v16h a = __builtin_shufflevector(alo, ahi, 0,1,2,3,4,5,6,7,8,9,10,11,12,13,14,15);
        int f = (k * 4 + nt) * 2 + kt;
        v16h bb = *(const v16h*)(w1p + (f << 9) + (lane << 4));
        acc = __builtin_amdgcn_wmma_f32_16x16x32_f16(false, a, false, bb,
                                                     (short)0, acc, false, false);
      }
      #pragma unroll
      for (int v = 0; v < 8; ++v)
        sm_hk[mt * 16 + v + 8 * hi][nt * 16 + nl] = (_Float16)acc[v];
    }
    __syncthreads();

    // attention dot products for head k
    if (t < NNODES) {
      const float* ps = as1 + k * HID;
      const float* pd = ad1 + k * HID;
      float vs = 0.f, vd = 0.f;
      for (int c = 0; c < HID; ++c) {
        float hv = (float)sm_hk[t][c];
        vs += hv * ps[c];
        vd += hv * pd[c];
      }
      sm_as[t] = vs; sm_ad[t] = vd;
    }
    __syncthreads();

    // segment softmax over 16 edges + self loop
    if (t < NNODES) {
      int d = t;
      float adv = sm_ad[d];
      float al[17];
      float mx = -3.4e38f;
      #pragma unroll
      for (int e = 0; e < 17; ++e) {
        int s = (e < DEG) ? sm_src[d * DEG + e] : d;
        float a = sm_as[s] + adv;
        a = (a > 0.f) ? a : (LEAKYF * a);
        al[e] = a;
        mx = fmaxf(mx, a);
      }
      float sum = 0.f;
      #pragma unroll
      for (int e = 0; e < 17; ++e) { al[e] = __expf(al[e] - mx); sum += al[e]; }
      float inv = 1.f / (sum + 1e-16f);
      #pragma unroll
      for (int e = 0; e < 17; ++e) sm_w[d][e] = al[e] * inv;
    }
    __syncthreads();

    // weighted scatter-aggregate + bias + ReLU -> out1[:, k*64 : k*64+64]
    {
      int d  = t >> 3;
      int cg = (t & 7) * 8;
      float accv[8] = {0.f, 0.f, 0.f, 0.f, 0.f, 0.f, 0.f, 0.f};
      #pragma unroll
      for (int e = 0; e < 17; ++e) {
        int   s = (e < DEG) ? sm_src[d * DEG + e] : d;
        float w = sm_w[d][e];
        h8 hv = *(const h8*)&sm_hk[s][cg];
        #pragma unroll
        for (int i = 0; i < 8; ++i) accv[i] += w * (float)hv[i];
      }
      #pragma unroll
      for (int i = 0; i < 8; ++i) {
        float o = accv[i] + b1[k * HID + cg + i];
        o = o > 0.f ? o : 0.f;
        sm_o1[d][k * HID + cg + i] = (_Float16)o;
      }
    }
    __syncthreads();
  }

  // ======================= GEMM2: H2 = out1 @ W2 ==========================
  {
    v8f acc = {};
    #pragma unroll
    for (int kt = 0; kt < 16; ++kt) {
      const h8 alo = *(const h8*)&sm_o1[m][kt * 32 + 8 * hi];
      const h8 ahi = *(const h8*)&sm_o1[m][kt * 32 + 16 + 8 * hi];
      v16h a = __builtin_shufflevector(alo, ahi, 0,1,2,3,4,5,6,7,8,9,10,11,12,13,14,15);
      int f = nt * 16 + kt;
      v16h bb = *(const v16h*)(w2p + (f << 9) + (lane << 4));
      acc = __builtin_amdgcn_wmma_f32_16x16x32_f16(false, a, false, bb,
                                                   (short)0, acc, false, false);
    }
    #pragma unroll
    for (int v = 0; v < 8; ++v)
      sm_hk[mt * 16 + v + 8 * hi][nt * 16 + nl] = (_Float16)acc[v];
  }
  __syncthreads();

  // ---- layer 2 attention --------------------------------------------------
  if (t < NNODES) {
    float vs = 0.f, vd = 0.f;
    for (int c = 0; c < HID; ++c) {
      float hv = (float)sm_hk[t][c];
      vs += hv * as2[c];
      vd += hv * ad2[c];
    }
    sm_as[t] = vs; sm_ad[t] = vd;
  }
  __syncthreads();

  if (t < NNODES) {
    int d = t;
    float adv = sm_ad[d];
    float al[17];
    float mx = -3.4e38f;
    #pragma unroll
    for (int e = 0; e < 17; ++e) {
      int s = (e < DEG) ? sm_src[d * DEG + e] : d;
      float a = sm_as[s] + adv;
      a = (a > 0.f) ? a : (LEAKYF * a);
      al[e] = a;
      mx = fmaxf(mx, a);
    }
    float sum = 0.f;
    #pragma unroll
    for (int e = 0; e < 17; ++e) { al[e] = __expf(al[e] - mx); sum += al[e]; }
    float inv = 1.f / (sum + 1e-16f);
    #pragma unroll
    for (int e = 0; e < 17; ++e) sm_w[d][e] = al[e] * inv;
  }
  __syncthreads();

  // ---- layer 2 aggregate + bias (no ReLU) -> out2 (f32) ------------------
  {
    int d  = t >> 3;
    int cg = (t & 7) * 8;
    float accv[8] = {0.f, 0.f, 0.f, 0.f, 0.f, 0.f, 0.f, 0.f};
    #pragma unroll
    for (int e = 0; e < 17; ++e) {
      int   s = (e < DEG) ? sm_src[d * DEG + e] : d;
      float w = sm_w[d][e];
      h8 hv = *(const h8*)&sm_hk[s][cg];
      #pragma unroll
      for (int i = 0; i < 8; ++i) accv[i] += w * (float)hv[i];
    }
    #pragma unroll
    for (int i = 0; i < 8; ++i)
      sm_o2[d][cg + i] = accv[i] + b2[cg + i];
  }
  __syncthreads();

  // ---- per-node head GEMV: logits[b,n,:] = out2[n] @ headW[n] + headb[n] --
  for (int idx = t; idx < NNODES * NCLASS; idx += 512) {
    int n = idx / NCLASS;
    int c = idx % NCLASS;
    float accv = headb[n * NCLASS + c];
    const float* wr = headW + (size_t)n * HID * NCLASS + c;
    #pragma unroll 8
    for (int d = 0; d < HID; ++d)
      accv += sm_o2[n][d] * wr[d * NCLASS];
    out[(size_t)b * NNODES * NCLASS + idx] = accv;
  }
}

// ---------------------------------------------------------------------------
extern "C" void kernel_launch(void* const* d_in, const int* in_sizes, int n_in,
                              void* d_out, int out_size, void* d_ws, size_t ws_size,
                              hipStream_t stream) {
  const int*   x    = (const int*)d_in[0];
  const int*   eidx = (const int*)d_in[1];
  const float* emb  = (const float*)d_in[2];
  const float* W1   = (const float*)d_in[3];
  const float* as1  = (const float*)d_in[4];
  const float* ad1  = (const float*)d_in[5];
  const float* b1   = (const float*)d_in[6];
  const float* W2   = (const float*)d_in[7];
  const float* as2  = (const float*)d_in[8];
  const float* ad2  = (const float*)d_in[9];
  const float* b2   = (const float*)d_in[10];
  const float* hW   = (const float*)d_in[11];
  const float* hb   = (const float*)d_in[12];

  _Float16* wp = (_Float16*)d_ws;   // 128 KB: packed f16 W1 + W2 fragments

  pack_weights<<<256, 256, 0, stream>>>(W1, W2, wp);
  gat_fused<<<256, 512, 0, stream>>>(x, eidx, emb, as1, ad1, b1,
                                     as2, ad2, b2, hW, hb, wp, (float*)d_out);
}
